// AttentionHead_76965813944883
// MI455X (gfx1250) — matrix operand order repaired and checked by reading, not compiled, not run on hardware
//
#include <hip/hip_runtime.h>
#include <hip/hip_bf16.h>

// ---------------------------------------------------------------------------
// Single-head causal attention for MI455X (gfx1250, wave32, WMMA + TDM).
//   B=8, T=2048, D=1024, H=64
// Pipeline:
//   1) wtrans:   W[q,k,v] (1024x64 f32) -> Wt (64x1024 f16)  [B-operand layout]
//   2) qkv_proj: fused Q/K/V projection; Wt chunk staged per-block by the
//                Tensor Data Mover (double-buffered), per-wave A pipelined,
//                v_wmma_f32_16x16x32_f16 accumulation
//   3) attn:     flash attention; K/Vt tiles staged by TDM (double-buffered,
//                s_wait_tensorcnt), online softmax, WMMA for QK^T and P@V
// ---------------------------------------------------------------------------

typedef __attribute__((ext_vector_type(16))) _Float16     v16h;
typedef __attribute__((ext_vector_type(8)))  _Float16     v8h;
typedef __attribute__((ext_vector_type(8)))  float        v8f;
typedef __attribute__((ext_vector_type(4)))  float        v4f;
typedef __attribute__((ext_vector_type(4)))  unsigned int u32x4;
typedef __attribute__((ext_vector_type(8)))  unsigned int u32x8;

#define SEQ_LEN  2048
#define D_EMBED  1024
#define HEAD_DIM 64
#define BATCH    8

__device__ __forceinline__ v16h pack16(v8h a, v8h b) {
    v16h r;
#pragma unroll
    for (int i = 0; i < 8; ++i) { r[i] = a[i]; r[8 + i] = b[i]; }
    return r;
}

// ---------------------------------------------------------------------------
// TDM: one 2D-tile DMA (global -> LDS), f16 elements.
// Descriptor per CDNA5 ISA.
//   pad_code: pad_interval code (pads after 1<<(code+1) DWORDs stored),
//   pad_amount fixed at 4 DWORDs (8 halves).
// Tracked by TENSORcnt; in-order per wave.
// ---------------------------------------------------------------------------
__device__ __forceinline__ void tdm_load_tile_f16(unsigned int lds_off,
                                                  const void* tile_gaddr,
                                                  unsigned int tensor_d0,
                                                  unsigned int tensor_d1,
                                                  unsigned int tile_d0,
                                                  unsigned int tile_d1,
                                                  unsigned int stride0,
                                                  unsigned int pad_code) {
    unsigned long long ga = (unsigned long long)(uintptr_t)tile_gaddr;
    u32x4 g0;
    g0[0] = 1u;                                           // count=1, user mode
    g0[1] = lds_off;                                      // LDS byte address
    g0[2] = (unsigned int)ga;                             // global_addr[31:0]
    g0[3] = (unsigned int)((ga >> 32) & 0x01ffffffu)      // global_addr[56:32]
            | (2u << 30);                                 // type = 2 ("image")
    u32x8 g1;
    g1[0] = (1u << 16)            // data_size = 1 -> 2 bytes
          | (1u << 20)            // pad_enable
          | (pad_code << 22)      // pad_interval
          | (3u << 25);           // pad_amount: 3+1 = 4 DWORDs = 8 halves
    g1[1] = (tensor_d0 & 0xffffu) << 16;                  // abar=0 | dim0.lo
    g1[2] = (tensor_d0 >> 16) | ((tensor_d1 & 0xffffu) << 16);
    g1[3] = (tensor_d1 >> 16) | (tile_d0 << 16);
    g1[4] = tile_d1;                                      // tile_dim2 = 0
    g1[5] = stride0;                                      // dim0_stride[31:0]
    g1[6] = 0u;                                           // stride0.hi | stride1.lo
    g1[7] = 0u;
    asm volatile("tensor_load_to_lds %0, %1" :: "s"(g0), "s"(g1) : "memory");
}

// ---------------------------------------------------------------------------
// Kernel 1: transpose W (1024x64 f32, row-major) -> Wt (64x1024 f16)
// ---------------------------------------------------------------------------
__global__ void wtrans_kernel(const float* __restrict__ Wq,
                              const float* __restrict__ Wk,
                              const float* __restrict__ Wv,
                              _Float16* __restrict__ WtQ,
                              _Float16* __restrict__ WtK,
                              _Float16* __restrict__ WtV) {
    int idx = blockIdx.x * 256 + threadIdx.x;   // 3 * 65536 total
    int m   = idx >> 16;
    int r   = idx & 65535;
    int n   = r & 63;        // head dim
    int k   = r >> 6;        // embed dim
    const float* W  = (m == 0) ? Wq : (m == 1) ? Wk : Wv;
    _Float16*    Wt = (m == 0) ? WtQ : (m == 1) ? WtK : WtV;
    Wt[(size_t)n * D_EMBED + k] = (_Float16)W[(size_t)k * HEAD_DIM + n];
}

// ---------------------------------------------------------------------------
// Kernel 2: fused QKV projection.
//   grid = (B*T)/128 blocks, 256 threads (8 waves); wave w -> 16 rows.
//   Wt k-chunk (3 x 64 x 32 f16) staged by TDM into LDS, double-buffered,
//   one chunk ahead; per-wave A tile (x rows, f32->f16) software-pipelined.
// ---------------------------------------------------------------------------
__global__ void __launch_bounds__(256)
qkv_proj_kernel(const float* __restrict__ x,
                const _Float16* __restrict__ WtQ,
                const _Float16* __restrict__ WtK,
                const _Float16* __restrict__ WtV,
                const float* __restrict__ bq,
                const float* __restrict__ bk,
                const float* __restrict__ bv,
                _Float16* __restrict__ Qo,
                _Float16* __restrict__ Ko,
                _Float16* __restrict__ Vto) {
    // TDM rows: 32 halves + 8 pad -> stride 40 halves (16B-aligned, no conflicts)
    __shared__ __align__(16) _Float16 WtS[2][3][64][40];   // 30 KB

    const int wave = threadIdx.x >> 5;
    const int lane = threadIdx.x & 31;
    const int lo   = lane & 15;
    const int hi   = lane >> 4;

    const int g0   = blockIdx.x * 128 + wave * 16;
    const int arow = g0 + lo;

    v8f acc[3][4];
#pragma unroll
    for (int m = 0; m < 3; ++m)
#pragma unroll
        for (int c = 0; c < 4; ++c) acc[m][c] = (v8f)0.0f;

    const float* xr = x + (size_t)arow * D_EMBED;
    const _Float16* const Wts[3] = {WtQ, WtK, WtV};

    // A tile 16x32 f16 at chunk k0: lane row lo, K = {hi*8+e, 16+hi*8+e}
    auto load_a = [&](int k0) -> v16h {
        const int kb = hi * 8;
        v4f f0 = *(const v4f*)(xr + k0 + kb);
        v4f f1 = *(const v4f*)(xr + k0 + kb + 4);
        v4f f2 = *(const v4f*)(xr + k0 + 16 + kb);
        v4f f3 = *(const v4f*)(xr + k0 + 16 + kb + 4);
        v16h a;
#pragma unroll
        for (int i = 0; i < 4; ++i) {
            a[i]      = (_Float16)f0[i];
            a[4 + i]  = (_Float16)f1[i];
            a[8 + i]  = (_Float16)f2[i];
            a[12 + i] = (_Float16)f3[i];
        }
        return a;
    };

    // prologue: A for chunk 0; wave 0 issues TDM triple for chunk 0 -> buf 0
    v16h a_cur = load_a(0);
    if (wave == 0) {
#pragma unroll
        for (int m = 0; m < 3; ++m)
            tdm_load_tile_f16((unsigned int)(uintptr_t)&WtS[0][m][0][0],
                              Wts[m], D_EMBED, HEAD_DIM, 32, HEAD_DIM,
                              D_EMBED, 3u /* pad every 16 DW = one 32h row */);
    }

    for (int ic = 0, k0 = 0; k0 < D_EMBED; ++ic, k0 += 32) {
        const int  cur  = ic & 1;
        const bool more = (k0 + 32) < D_EMBED;

        v16h a_next = a_cur;
        if (more) a_next = load_a(k0 + 32);   // overlaps with barriers/TDM wait

        __syncthreads();   // all waves done with buffer 'nxt' (chunk ic-1)
        if (wave == 0) {
            if (more) {
                const int nxt = cur ^ 1;
#pragma unroll
                for (int m = 0; m < 3; ++m)
                    tdm_load_tile_f16((unsigned int)(uintptr_t)&WtS[nxt][m][0][0],
                                      Wts[m] + (k0 + 32), D_EMBED, HEAD_DIM,
                                      32, HEAD_DIM, D_EMBED, 3u);
                __builtin_amdgcn_s_wait_tensorcnt(3);   // oldest triple done
            } else {
                __builtin_amdgcn_s_wait_tensorcnt(0);
            }
        }
        __syncthreads();   // buffer 'cur' visible to all waves

        // B tiles from LDS: lane col c*16+lo, K = {hi*8+e, 16+hi*8+e}
#pragma unroll
        for (int m = 0; m < 3; ++m) {
#pragma unroll
            for (int c = 0; c < 4; ++c) {
                const _Float16* src = &WtS[cur][m][c * 16 + lo][hi * 8];
                v16h bm = pack16(*(const v8h*)src, *(const v8h*)(src + 16));
                acc[m][c] = __builtin_amdgcn_wmma_f32_16x16x32_f16(
                    false, a_cur, false, bm, (short)0, acc[m][c], false, false);
            }
        }
        a_cur = a_next;
    }

    // bias + store. C layout: row = g0 + r + 8*hi, col = c*16 + lo
    const int b  = g0 >> 11;
    const int t0 = g0 & (SEQ_LEN - 1);
#pragma unroll
    for (int c = 0; c < 4; ++c) {
        const int   n   = c * 16 + lo;
        const float bqv = bq[n], bkv = bk[n], bvv = bv[n];
#pragma unroll
        for (int r = 0; r < 8; ++r) {
            const int grow = g0 + r + 8 * hi;
            const int t    = t0 + r + 8 * hi;
            Qo[(size_t)grow * HEAD_DIM + n] = (_Float16)(acc[0][c][r] + bqv);
            Ko[(size_t)grow * HEAD_DIM + n] = (_Float16)(acc[1][c][r] + bkv);
            Vto[(size_t)b * HEAD_DIM * SEQ_LEN + (size_t)n * SEQ_LEN + t] =
                (_Float16)(acc[2][c][r] + bvv);
        }
    }
}

// ---------------------------------------------------------------------------
// Kernel 3: flash attention with TDM-staged, double-buffered K/Vt tiles.
//   grid = B * (T/128) blocks, 256 threads (8 waves); wave w -> 16 queries.
// ---------------------------------------------------------------------------
__global__ void __launch_bounds__(256)
attn_kernel(const _Float16* __restrict__ Q,
            const _Float16* __restrict__ K,
            const _Float16* __restrict__ Vt,
            float* __restrict__ out) {
    // TDM rows: 64 halves + 8 pad -> stride 72 halves (no bank conflicts)
    __shared__ __align__(16) _Float16 Ks [2][64][72];    // keys x headdim
    __shared__ __align__(16) _Float16 Vts[2][64][72];    // headdim x keys
    __shared__ __align__(16) _Float16 Ps [8][16][72];    // per-wave P scratch

    const int tid  = threadIdx.x;
    const int wave = tid >> 5;
    const int lane = tid & 31;
    const int lo   = lane & 15;
    const int hi   = lane >> 4;

    const int b     = blockIdx.x >> 4;           // 16 query blocks per batch
    const int q0    = (blockIdx.x & 15) * 128;
    const int qbase = q0 + wave * 16;

    const _Float16* Qb = Q  + (size_t)b * SEQ_LEN * HEAD_DIM;
    const _Float16* Kb = K  + (size_t)b * SEQ_LEN * HEAD_DIM;
    const _Float16* Vb = Vt + (size_t)b * HEAD_DIM * SEQ_LEN;

    // ---- load Q A-tiles once: 16 x 64 -> two v16h (K chunks of 32)
    v16h aq[2];
    {
        const _Float16* qr = Qb + (size_t)(qbase + lo) * HEAD_DIM + hi * 8;
        aq[0] = pack16(*(const v8h*)qr,        *(const v8h*)(qr + 16));
        aq[1] = pack16(*(const v8h*)(qr + 32), *(const v8h*)(qr + 48));
    }

    v8f o[4];
#pragma unroll
    for (int c = 0; c < 4; ++c) o[c] = (v8f)0.0f;
    float m[8], l[8];
#pragma unroll
    for (int r = 0; r < 8; ++r) { m[r] = -1.0e30f; l[r] = 0.0f; }

    const int   qtile_end = qbase + 15;
    const int   kend      = q0 + 127;
    const float sc        = 0.125f;   // 1/sqrt(64)

    // ---- prologue: wave 0 issues TDM for key block 0 into buffer 0
    if (wave == 0) {
        tdm_load_tile_f16((unsigned int)(uintptr_t)&Ks[0][0][0],  Kb,
                          HEAD_DIM, SEQ_LEN, HEAD_DIM, 64, HEAD_DIM, 4u);
        tdm_load_tile_f16((unsigned int)(uintptr_t)&Vts[0][0][0], Vb,
                          SEQ_LEN, HEAD_DIM, 64, HEAD_DIM, SEQ_LEN, 4u);
    }

    for (int ib = 0, kblk = 0; kblk <= kend; ++ib, kblk += 64) {
        const int cur = ib & 1;
        __syncthreads();   // everyone done computing from buffer nxt (iter ib-1)

        const bool have_next = (kblk + 64) <= kend;
        if (wave == 0) {
            if (have_next) {
                const int nxt = cur ^ 1;
                tdm_load_tile_f16((unsigned int)(uintptr_t)&Ks[nxt][0][0],
                                  Kb + (size_t)(kblk + 64) * HEAD_DIM,
                                  HEAD_DIM, SEQ_LEN, HEAD_DIM, 64, HEAD_DIM, 4u);
                tdm_load_tile_f16((unsigned int)(uintptr_t)&Vts[nxt][0][0],
                                  Vb + (kblk + 64),
                                  SEQ_LEN, HEAD_DIM, 64, HEAD_DIM, SEQ_LEN, 4u);
                __builtin_amdgcn_s_wait_tensorcnt(2);  // oldest pair (cur) done
            } else {
                __builtin_amdgcn_s_wait_tensorcnt(0);
            }
        }
        __syncthreads();   // buffer cur visible to all waves

        if (kblk <= qtile_end) {     // wave-uniform: EXEC stays all-ones
            // ---- S = Q K^T : 4 key sub-tiles x 2 K-chunks of headdim
            v8f s[4];
#pragma unroll
            for (int st = 0; st < 4; ++st) {
                v8f accs = (v8f)0.0f;
#pragma unroll
                for (int kc = 0; kc < 2; ++kc) {
                    const _Float16* kp = &Ks[cur][st * 16 + lo][kc * 32 + hi * 8];
                    v16h bm = pack16(*(const v8h*)kp, *(const v8h*)(kp + 16));
                    accs = __builtin_amdgcn_wmma_f32_16x16x32_f16(
                        false, aq[kc], false, bm, (short)0, accs, false, false);
                }
                s[st] = accs;
            }
            // ---- scale + causal mask
#pragma unroll
            for (int st = 0; st < 4; ++st) {
                const int key = kblk + st * 16 + lo;
#pragma unroll
                for (int r = 0; r < 8; ++r) {
                    const int q = qbase + r + 8 * hi;
                    s[st][r] = (key <= q) ? s[st][r] * sc : -1.0e30f;
                }
            }
            // ---- online softmax: row max via 16-lane butterfly
            float rm[8];
#pragma unroll
            for (int r = 0; r < 8; ++r)
                rm[r] = fmaxf(fmaxf(s[0][r], s[1][r]), fmaxf(s[2][r], s[3][r]));
#pragma unroll
            for (int msk = 1; msk <= 8; msk <<= 1)
#pragma unroll
                for (int r = 0; r < 8; ++r)
                    rm[r] = fmaxf(rm[r], __shfl_xor(rm[r], msk, 32));

            float corr[8], rs[8];
#pragma unroll
            for (int r = 0; r < 8; ++r) {
                const float mn = fmaxf(m[r], rm[r]);
                corr[r] = __expf(m[r] - mn);
                m[r]    = mn;
                rs[r]   = 0.0f;
            }
#pragma unroll
            for (int st = 0; st < 4; ++st)
#pragma unroll
                for (int r = 0; r < 8; ++r) {
                    const float p = __expf(s[st][r] - m[r]);
                    s[st][r] = p;
                    rs[r] += p;
                }
#pragma unroll
            for (int msk = 1; msk <= 8; msk <<= 1)
#pragma unroll
                for (int r = 0; r < 8; ++r)
                    rs[r] += __shfl_xor(rs[r], msk, 32);
#pragma unroll
            for (int r = 0; r < 8; ++r) l[r] = l[r] * corr[r] + rs[r];
#pragma unroll
            for (int c = 0; c < 4; ++c)
#pragma unroll
                for (int r = 0; r < 8; ++r) o[c][r] *= corr[r];

            // ---- P: C-layout -> per-wave LDS -> A-layout (DS in-order per wave)
#pragma unroll
            for (int st = 0; st < 4; ++st)
#pragma unroll
                for (int r = 0; r < 8; ++r)
                    Ps[wave][r + 8 * hi][st * 16 + lo] = (_Float16)s[st][r];
            asm volatile("" ::: "memory");

            // ---- O += P @ V
#pragma unroll
            for (int kc = 0; kc < 2; ++kc) {
                const _Float16* pp = &Ps[wave][lo][kc * 32 + hi * 8];
                v16h ap = pack16(*(const v8h*)pp, *(const v8h*)(pp + 16));
#pragma unroll
                for (int c = 0; c < 4; ++c) {
                    const _Float16* vp = &Vts[cur][c * 16 + lo][kc * 32 + hi * 8];
                    v16h bm = pack16(*(const v8h*)vp, *(const v8h*)(vp + 16));
                    o[c] = __builtin_amdgcn_wmma_f32_16x16x32_f16(
                        false, ap, false, bm, (short)0, o[c], false, false);
                }
            }
        }
    }

    // ---- normalize and store fp32 output
    float* ob = out + (size_t)b * SEQ_LEN * HEAD_DIM;
#pragma unroll
    for (int r = 0; r < 8; ++r) {
        const float inv = 1.0f / l[r];
        const int   q   = qbase + r + 8 * hi;
#pragma unroll
        for (int c = 0; c < 4; ++c)
            ob[(size_t)q * HEAD_DIM + c * 16 + lo] = o[c][r] * inv;
    }
}

// ---------------------------------------------------------------------------
extern "C" void kernel_launch(void* const* d_in, const int* in_sizes, int n_in,
                              void* d_out, int out_size, void* d_ws, size_t ws_size,
                              hipStream_t stream) {
    (void)in_sizes; (void)n_in; (void)out_size; (void)ws_size;
    const float* x  = (const float*)d_in[0];
    const float* Wq = (const float*)d_in[1];
    const float* bq = (const float*)d_in[2];
    const float* Wk = (const float*)d_in[3];
    const float* bk = (const float*)d_in[4];
    const float* Wv = (const float*)d_in[5];
    const float* bv = (const float*)d_in[6];
    float* out = (float*)d_out;

    // workspace layout (f16 elements), all offsets 256B aligned
    _Float16* ws  = (_Float16*)d_ws;
    _Float16* WtQ = ws;                                   //  64*1024
    _Float16* WtK = WtQ + HEAD_DIM * D_EMBED;
    _Float16* WtV = WtK + HEAD_DIM * D_EMBED;
    _Float16* Qh  = WtV + HEAD_DIM * D_EMBED;             // B*T*64 each
    _Float16* Kh  = Qh + (size_t)BATCH * SEQ_LEN * HEAD_DIM;
    _Float16* Vth = Kh + (size_t)BATCH * SEQ_LEN * HEAD_DIM;

    wtrans_kernel<<<(3 * D_EMBED * HEAD_DIM) / 256, 256, 0, stream>>>(
        Wq, Wk, Wv, WtQ, WtK, WtV);
    qkv_proj_kernel<<<(BATCH * SEQ_LEN) / 128, 256, 0, stream>>>(
        x, WtQ, WtK, WtV, bq, bk, bv, Qh, Kh, Vth);
    attn_kernel<<<BATCH * (SEQ_LEN / 128), 256, 0, stream>>>(Qh, Kh, Vth, out);
}
